// Hyper_encoder_G_78915729097356
// MI455X (gfx1250) — matrix operand order repaired and checked by reading, not compiled, not run on hardware
//
#include <hip/hip_runtime.h>
#include <hip/hip_bf16.h>

// ---------------------------------------------------------------------------
// MI455X (gfx1250) masked 3D-conv encoder.
// Convs = implicit GEMM on v_wmma_f32_16x16x32_bf16 (wave32 WMMA):
//   M = out channels (64 -> 4 tiles of 16, one per wave, 128-thr block)
//   N = 16*NT output voxels along W (NT=2 for stride-1 layers: A-reuse)
//   K = in channels (64 = 2 chunks of 32) x 27 taps (fully unrolled,
//       branch-free: LDS halo is zero-filled at grid boundaries)
// Halo staged in LDS via gfx1250 async global->LDS loads (ASYNCcnt).
// ---------------------------------------------------------------------------

typedef __attribute__((ext_vector_type(16))) __bf16 v16bf;
typedef __attribute__((ext_vector_type(8)))  __bf16 v8bf;
typedef __attribute__((ext_vector_type(8)))  float  v8f;

#define C_CH   64
#define G0     64
#define NVOX0  (G0*G0*G0)        // 262144
#define ASYNC_HALO 1

// ---- mask int -> float -----------------------------------------------------
__global__ void k_maskf(const int* __restrict__ m, float* __restrict__ mf, int n) {
    int i = blockIdx.x * blockDim.x + threadIdx.x;
    if (i < n) mf[i] = (float)m[i];
}

// ---- 2x2x2 max-pool of occupancy ------------------------------------------
__global__ void k_pool(const float* __restrict__ in, float* __restrict__ out, int Go) {
    int i = blockIdx.x * blockDim.x + threadIdx.x;
    int n = Go * Go * Go;
    if (i >= n) return;
    int w = i % Go, h = (i / Go) % Go, d = i / (Go * Go);
    int Gi = Go * 2;
    float mv = 0.f;
    #pragma unroll
    for (int dz = 0; dz < 2; ++dz)
        #pragma unroll
        for (int hz = 0; hz < 2; ++hz)
            #pragma unroll
            for (int wz = 0; wz < 2; ++wz)
                mv = fmaxf(mv, in[(((size_t)(2*d+dz) * Gi) + (2*h+hz)) * Gi + (2*w+wz)]);
    out[i] = mv;
}

// ---- weight repack: OIDHW f32 -> A-fragment-ordered bf16 -------------------
// Flat layout: [tap(27)][kc(2)][mt(4)][lane(32)][e(16)]
// 16-bit A 16x32 layout used here (contiguous k, same as B): lane = 16*half + m ; element e -> k =
//   half*16 + e
__global__ void k_repack(const float* __restrict__ w, __bf16* __restrict__ wf) {
    int idx = blockIdx.x * blockDim.x + threadIdx.x;
    if (idx >= 27 * 2 * 4 * 32 * 16) return;
    int e    = idx & 15;
    int lane = (idx >> 4) & 31;
    int mt   = (idx >> 9) & 3;
    int kc   = (idx >> 11) & 1;
    int tap  = idx >> 12;
    int m = lane & 15, half = lane >> 4;
    int k  = half * 16 + e;
    int co = mt * 16 + m;
    int ci = kc * 32 + k;
    wf[idx] = (__bf16)w[((size_t)co * 64 + ci) * 27 + tap];
}

// ---- pack x * m0 -> bf16, voxel-major [v][ci] ------------------------------
__global__ void k_pack(const float* __restrict__ x, const float* __restrict__ m0,
                       __bf16* __restrict__ act) {
    int i = blockIdx.x * blockDim.x + threadIdx.x;
    if (i >= C_CH * NVOX0) return;
    int v = i >> 6, c = i & 63;
    act[i] = (__bf16)(x[(size_t)c * NVOX0 + v] * m0[v]);
}

// ---- generic 3x3x3 conv, implicit GEMM on WMMA -----------------------------
// Grid: (Gout/(16*NT), Gout, Gout) ; block: 128 threads (4 waves).
template<int S, int NT, bool RELU, bool OUTF32>
__global__ __launch_bounds__(128)
void conv3x3x3_wmma(const __bf16* __restrict__ in,      // [Gin^3][64] bf16
                    const __bf16* __restrict__ wfrag,   // A-fragment layout
                    const float*  __restrict__ bias,    // [64]
                    const float*  __restrict__ maskO,   // [Gout^3]
                    __bf16* __restrict__ outB,          // [Gout^3][64] bf16
                    float*  __restrict__ outF,          // NCDHW f32 (final)
                    int Gin, int Gout)
{
    constexpr int WH   = 16 * NT * S + 2;      // input-w halo width per block
    constexpr int LDSP = 72;                   // padded ci pitch (rows 16B aligned)
    __shared__ __align__(16) __bf16 lds[9 * WH * LDSP];

    const int w0 = blockIdx.x * (16 * NT);
    const int h0 = blockIdx.y;
    const int d0 = blockIdx.z;

    // ---- halo fill: async global->LDS, zero-store for out-of-range chunks --
    // Each 16B chunk is owned by exactly one thread (no write races).
    const unsigned ldsBase = (unsigned)(unsigned long long)(&lds[0]);
    const int nchunk = 9 * WH * 8;             // 16B chunks
    for (int i = threadIdx.x; i < nchunk; i += 128) {
        const int cc = i & 7;
        const int wi = (i >> 3) % WH;
        const int rr = i / (8 * WH);           // rd*3+rh
        const int rd = rr / 3, rh = rr % 3;
        const int id = d0 * S + rd - 1;
        const int ih = h0 * S + rh - 1;
        const int iw = w0 * S + wi - 1;
        const int loff = (rr * WH + wi) * LDSP + cc * 8;   // elements
        const bool inr = (unsigned)id < (unsigned)Gin &&
                         (unsigned)ih < (unsigned)Gin &&
                         (unsigned)iw < (unsigned)Gin;
#if ASYNC_HALO
        if (inr) {
            const size_t vin = ((size_t)id * Gin + ih) * Gin + iw;
            const __bf16* gp = in + vin * 64 + cc * 8;
            const unsigned lpos = ldsBase + (unsigned)loff * 2;
            asm volatile("global_load_async_to_lds_b128 %0, %1, off"
                         :: "v"(lpos), "v"((unsigned long long)gp)
                         : "memory");
        } else {
            v8bf z;
            #pragma unroll
            for (int j = 0; j < 8; ++j) z[j] = (__bf16)0.0f;
            *(v8bf*)(&lds[loff]) = z;
        }
#else
        v8bf val;
        #pragma unroll
        for (int j = 0; j < 8; ++j) val[j] = (__bf16)0.0f;
        if (inr) {
            const size_t vin = ((size_t)id * Gin + ih) * Gin + iw;
            val = *(const v8bf*)(in + vin * 64 + cc * 8);
        }
        *(v8bf*)(&lds[loff]) = val;
#endif
    }
#if ASYNC_HALO
    asm volatile("s_wait_asynccnt 0x0" ::: "memory");
#endif
    __syncthreads();

    const int lane = threadIdx.x & 31;
    const int mt   = threadIdx.x >> 5;         // M-tile per wave
    const int nIdx = lane & 15;
    const int half = lane >> 4;

    // Per-lane invariant bases: weight base for this (mt, lane), LDS base for
    // this lane's column/half. All tap offsets become literal immediates.
    const __bf16* wbase = wfrag + (size_t)(((mt * 32) + lane) << 4);
    const int ldsCol = nIdx * S * LDSP + half * 16;

    v8f acc[NT];
    #pragma unroll
    for (int j = 0; j < NT; ++j)
        #pragma unroll
        for (int r = 0; r < 8; ++r) acc[j][r] = 0.0f;

    // Branch-free: out-of-grid taps read zero-filled LDS rows (contribute 0).
    #pragma unroll
    for (int t = 0; t < 27; ++t) {
        const int td = t / 9, th = (t / 3) % 3, tw = t % 3;
        #pragma unroll
        for (int kc = 0; kc < 2; ++kc) {
            // A: pre-swizzled weights -> one 32B load per lane, reused NT times
            v16bf a = *(const v16bf*)(wbase + ((t * 2 + kc) << 9));
            #pragma unroll
            for (int j = 0; j < NT; ++j) {
                const int rowOff =
                    (((td * 3 + th) * WH + tw) + 16 * j * S) * LDSP + kc * 32;
                v16bf b = *(const v16bf*)(&lds[ldsCol + rowOff]);
                acc[j] = __builtin_amdgcn_wmma_f32_16x16x32_bf16(
                             false, a, false, b, (short)0, acc[j], false, false);
            }
        }
    }

    // ---- epilogue: bias, mask, relu, store ----
    const int coBase = mt * 16 + half * 8;     // C/D layout: VGPR r -> M = r + 8*half
    #pragma unroll
    for (int j = 0; j < NT; ++j) {
        const int vox = ((d0 * Gout) + h0) * Gout + w0 + 16 * j + nIdx;
        const float mval = maskO[vox];
        if constexpr (OUTF32) {
            const size_t plane = (size_t)Gout * Gout * Gout;
            #pragma unroll
            for (int r = 0; r < 8; ++r) {
                float v = (acc[j][r] + bias[coBase + r]) * mval;
                if (RELU) v = fmaxf(v, 0.0f);
                outF[(size_t)(coBase + r) * plane + vox] = v;
            }
        } else {
            v8bf ov;
            #pragma unroll
            for (int r = 0; r < 8; ++r) {
                float v = (acc[j][r] + bias[coBase + r]) * mval;
                if (RELU) v = fmaxf(v, 0.0f);
                ov[r] = (__bf16)v;
            }
            *(v8bf*)(outB + (size_t)vox * 64 + coBase) = ov;
        }
    }
}

// ---------------------------------------------------------------------------
extern "C" void kernel_launch(void* const* d_in, const int* in_sizes, int n_in,
                              void* d_out, int out_size, void* d_ws, size_t ws_size,
                              hipStream_t stream) {
    (void)in_sizes; (void)n_in; (void)out_size; (void)ws_size;

    const float* x     = (const float*)d_in[0];
    const int*   maskI = (const int*)  d_in[1];
    const float* w[5]  = { (const float*)d_in[2], (const float*)d_in[4],
                           (const float*)d_in[6], (const float*)d_in[8],
                           (const float*)d_in[10] };
    const float* b[5]  = { (const float*)d_in[3], (const float*)d_in[5],
                           (const float*)d_in[7], (const float*)d_in[9],
                           (const float*)d_in[11] };
    float* out = (float*)d_out;

    char* ws = (char*)d_ws;
    size_t off = 0;
    auto carve = [&](size_t bytes) {
        void* p = ws + off;
        off = (off + bytes + 255) & ~(size_t)255;
        return p;
    };
    float*  m0f = (float*)carve((size_t)NVOX0 * 4);
    float*  m1f = (float*)carve((size_t)32768 * 4);
    float*  m2f = (float*)carve((size_t)4096 * 4);
    __bf16* wf[5];
    for (int l = 0; l < 5; ++l) wf[l] = (__bf16*)carve((size_t)110592 * 2);
    __bf16* actA = (__bf16*)carve((size_t)NVOX0 * 64 * 2);   // 32 MB
    __bf16* actB = (__bf16*)carve((size_t)NVOX0 * 64 * 2);   // 32 MB

    // masks
    k_maskf<<<NVOX0 / 256, 256, 0, stream>>>(maskI, m0f, NVOX0);
    k_pool <<<32768 / 256, 256, 0, stream>>>(m0f, m1f, 32);
    k_pool <<<4096  / 256, 256, 0, stream>>>(m1f, m2f, 16);

    // weights -> bf16 A-fragments
    for (int l = 0; l < 5; ++l)
        k_repack<<<110592 / 256, 256, 0, stream>>>(w[l], wf[l]);

    // x * m0 -> bf16 [v][ci]
    k_pack<<<(C_CH * NVOX0) / 256, 256, 0, stream>>>(x, m0f, actA);

    // layer 1: conv0 s1 NT2, relu, mask m0      (64^3)
    conv3x3x3_wmma<1, 2, true,  false><<<dim3(2, 64, 64), 128, 0, stream>>>(
        actA, wf[0], b[0], m0f, actB, nullptr, 64, 64);
    // layer 2: conv1 s1 NT2, mask m0            (64^3)
    conv3x3x3_wmma<1, 2, false, false><<<dim3(2, 64, 64), 128, 0, stream>>>(
        actB, wf[1], b[1], m0f, actA, nullptr, 64, 64);
    // layer 3: down0 s2 NT1, relu, mask m1      (64^3 -> 32^3)
    conv3x3x3_wmma<2, 1, true,  false><<<dim3(2, 32, 32), 128, 0, stream>>>(
        actA, wf[2], b[2], m1f, actB, nullptr, 64, 32);
    // layer 4: conv2 s1 NT2, mask m1            (32^3)
    conv3x3x3_wmma<1, 2, false, false><<<dim3(1, 32, 32), 128, 0, stream>>>(
        actB, wf[3], b[3], m1f, actA, nullptr, 32, 32);
    // layer 5: down1 s2 NT1, mask m2, f32 NCDHW out (32^3 -> 16^3)
    conv3x3x3_wmma<2, 1, false, true ><<<dim3(1, 16, 16), 128, 0, stream>>>(
        actA, wf[4], b[4], m2f, nullptr, out, 32, 16);
}
